// TransPooling_58213986730024
// MI455X (gfx1250) — matrix operand order repaired and checked by reading, not compiled, not run on hardware
//
#include <hip/hip_runtime.h>

// Problem sizes (fixed by the reference)
#define BDIM 8
#define CDIM 256
#define NDIM 4096

typedef __attribute__((ext_vector_type(16))) __bf16 v16bf;
typedef __attribute__((ext_vector_type(8)))  float  v8f;
typedef __attribute__((ext_vector_type(4)))  unsigned int v4u;
typedef __attribute__((ext_vector_type(8)))  unsigned int v8u;

__device__ __forceinline__ unsigned short f2bf(float f) {
    // round-to-nearest-even f32 -> bf16
    unsigned int u = __float_as_uint(f);
    u += 0x7FFFu + ((u >> 16) & 1u);
    return (unsigned short)(u >> 16);
}

// ---------------------------------------------------------------------------
// TDM: DMA one 64-row x 256-col bf16 tile (row stride 256 elems in memory)
// into LDS, inserting 16B of padding after every 512B row so the LDS layout
// has stride 264 ushorts (matches the WMMA fragment addressing below).
//   D# group0: count=1 | lds_addr | global_addr[56:0] | type=2
//   D# group1: data_size=2B, pad_enable, pad_interval=6 (512B), pad_amount=3
//              (4 dwords), tensor_dim0=256, tensor_dim1=4096, tile=256x64,
//              tensor_dim0_stride=256
// Issued by one wave; completion tracked on TENSORcnt.
// ---------------------------------------------------------------------------
__device__ __forceinline__ void tdm_load_tile(unsigned lds_addr,
                                              unsigned long long gaddr)
{
    v4u g0;
    g0[0] = 1u;                                   // count=1 (valid descriptor)
    g0[1] = lds_addr;                             // LDS dest byte address
    g0[2] = (unsigned)gaddr;                      // global_addr[31:0]
    g0[3] = (unsigned)((gaddr >> 32) & 0x01FFFFFFull) | (2u << 30); // type=2
    v8u g1;
    g1[0] = (1u << 16)        // data_size = 1 -> 2 bytes
          | (1u << 20)        // pad_enable
          | (6u << 22)        // pad_interval: 2^(6+1) = 128 dwords = 512B
          | (3u << 25);       // pad_amount: 3+1 = 4 dwords = 16B
    g1[1] = (256u << 16);     // tensor_dim0[15:0]=256 (atomic_barrier_addr=0)
    g1[2] = (4096u << 16);    // tensor_dim0[31:16]=0, tensor_dim1[15:0]=4096
    g1[3] = (256u << 16);     // tensor_dim1[31:16]=0, tile_dim0=256
    g1[4] = 64u;              // tile_dim1=64, tile_dim2=0
    g1[5] = 256u;             // tensor_dim0_stride[31:0]=256
    g1[6] = 0u;               // stride0[47:32]=0, tensor_dim1_stride lo=0
    g1[7] = 0u;
    asm volatile("tensor_load_to_lds %0, %1" :: "s"(g0), "s"(g1) : "memory");
}

// ---------------------------------------------------------------------------
// Kernel 1: Q/K projections.  Q[n,d] = sum_c X[c,n]*Wq[d,c] (and K with Wk).
// Block: 256 thr (8 waves), 64 n-rows x 256 d-cols.  Wave w: mat=w>>2 (Q/K),
// n-tile i=w&3, 16 d-tiles of 16x16x32 bf16 WMMA accumulated over C=256.
// ---------------------------------------------------------------------------
__global__ __launch_bounds__(256) void proj_kernel(
    const float* __restrict__ X, const float* __restrict__ Wq,
    const float* __restrict__ Wk, unsigned short* __restrict__ Qb,
    unsigned short* __restrict__ Kb)
{
    constexpr int XS = 34;   // lds_x row stride (bf16 elems), padded
    constexpr int WS = 34;   // lds_w row stride
    __shared__ unsigned short lds_x[64 * XS];     // X^T chunk: [n][c]
    __shared__ unsigned short lds_wq[256 * WS];   // Wq chunk:  [d][c]
    __shared__ unsigned short lds_wk[256 * WS];   // Wk chunk:  [d][c]

    const int tid = threadIdx.x;
    const int b   = blockIdx.y;
    const int n0  = blockIdx.x * 64;
    const int w    = tid >> 5, lane = tid & 31;
    const int mat  = w >> 2,  i    = w & 3;
    const int ln   = lane & 15;
    const bool hi  = lane >= 16;

    v8f acc[16];
    #pragma unroll
    for (int t = 0; t < 16; ++t) acc[t] = v8f{0.f,0.f,0.f,0.f,0.f,0.f,0.f,0.f};

    for (int c0 = 0; c0 < CDIM; c0 += 32) {
        // stage X^T chunk: 32c x 64n, coalesced in n, convert to bf16
        #pragma unroll
        for (int it = 0; it < 8; ++it) {
            int idx = tid + it * 256;                 // 0..2047
            int cl = idx >> 6, nl = idx & 63;
            float x = X[((size_t)b * CDIM + c0 + cl) * NDIM + n0 + nl];
            lds_x[nl * XS + cl] = f2bf(x);
        }
        // stage W chunks as [d][c] (pairs contiguous for B fragments)
        #pragma unroll
        for (int it = 0; it < 16; ++it) {
            int pid = tid + it * 256;                 // 0..4095 pairs
            int d = pid >> 4, c2 = (pid & 15) * 2;
            float2 q2 = *(const float2*)&Wq[d * CDIM + c0 + c2];
            float2 k2 = *(const float2*)&Wk[d * CDIM + c0 + c2];
            *(unsigned int*)&lds_wq[d * WS + c2] =
                (unsigned int)f2bf(q2.x) | ((unsigned int)f2bf(q2.y) << 16);
            *(unsigned int*)&lds_wk[d * WS + c2] =
                (unsigned int)f2bf(k2.x) | ((unsigned int)f2bf(k2.y) << 16);
        }
        __syncthreads();

        // A fragment: 16x32 bf16, rows = n-tile i
        union { v16bf v; unsigned int u[8]; } a;
        const unsigned short* abase = &lds_x[(i * 16 + ln) * XS];
        #pragma unroll
        for (int p = 0; p < 8; ++p) {
            int k = (hi ? 16 : 0) + 2 * p;
            a.u[p] = *(const unsigned int*)(abase + k);
        }
        const unsigned short* wlds = mat ? lds_wk : lds_wq;
        #pragma unroll
        for (int dt = 0; dt < 16; ++dt) {
            union { v16bf v; unsigned int u[8]; } bf;
            const unsigned short* bbase = &wlds[(dt * 16 + ln) * WS + (hi ? 16 : 0)];
            #pragma unroll
            for (int p = 0; p < 8; ++p)
                bf.u[p] = *(const unsigned int*)(bbase + 2 * p);
            acc[dt] = __builtin_amdgcn_wmma_f32_16x16x32_bf16(
                false, a.v, false, bf.v, (short)0, acc[dt], false, false);
        }
        __syncthreads();
    }

    unsigned short* O = mat ? Kb : Qb;
    #pragma unroll
    for (int dt = 0; dt < 16; ++dt) {
        #pragma unroll
        for (int j = 0; j < 8; ++j) {
            int n_ = n0 + i * 16 + j + (hi ? 8 : 0);
            int d_ = dt * 16 + ln;
            O[((size_t)b * NDIM + n_) * CDIM + d_] = f2bf(acc[dt][j]);
        }
    }
}

// ---------------------------------------------------------------------------
// Kernel 2: flash-softmax row stats + diagonal.  Block owns 64 Q rows
// (A-fragments hoisted into registers), loops 64 K tiles double-buffered in
// LDS via TDM DMA; wave (i = n-tile, jw = 32-col half) keeps a 16x32 S tile
// in 2 WMMA accumulators; online max/sum across the full row.
// ---------------------------------------------------------------------------
__global__ __launch_bounds__(256) void attn_diag_kernel(
    const unsigned short* __restrict__ Qb,
    const unsigned short* __restrict__ Kb,
    float* __restrict__ diag)
{
    constexpr int QS = 264;  // row stride (bf16 elems) incl. TDM padding
    __shared__ unsigned short lds_q[64 * QS];
    __shared__ unsigned short lds_k[2][64 * QS];   // double buffer (TDM dest)
    __shared__ float wmaxs[2][64];
    __shared__ float wsums[2][64];
    __shared__ float draw[64];      // raw diagonal logits

    const int tid = threadIdx.x;
    const int b   = blockIdx.y;
    const int n0  = blockIdx.x * 64;
    const int w   = tid >> 5, lane = tid & 31;
    const int jw  = w >> 2,  i    = w & 3;
    const int ln  = lane & 15;
    const bool hi = lane >= 16;

    const unsigned short* kbase = Kb + (size_t)b * NDIM * CDIM;

    // stage resident Q tile (64 x 256 bf16) via uint4
    const uint4* qg = (const uint4*)(Qb + ((size_t)b * NDIM + n0) * CDIM);
    #pragma unroll
    for (int it = 0; it < 8; ++it) {
        int idx = tid + it * 256;          // 2048 uint4
        int r = idx >> 5, c4 = idx & 31;
        uint4 v = qg[r * 32 + c4];
        *(uint4*)&lds_q[r * QS + c4 * 8] = v;
    }

    // kick off DMA of K tile 0 into buffer 0 (one wave issues; TENSORcnt)
    if (w == 0)
        tdm_load_tile((unsigned)(uintptr_t)(void*)&lds_k[0][0],
                      (unsigned long long)(uintptr_t)kbase);
    __syncthreads();   // Q tile visible

    // hoist the 8 loop-invariant A fragments (Q rows of n-tile i) to VGPRs
    union Frag { v16bf v; unsigned int u[8]; };
    Frag a[8];
    #pragma unroll
    for (int kc = 0; kc < 8; ++kc) {
        const unsigned short* abase = &lds_q[(i * 16 + ln) * QS + kc * 32];
        #pragma unroll
        for (int p = 0; p < 8; ++p) {
            int k = (hi ? 16 : 0) + 2 * p;
            a[kc].u[p] = *(const unsigned int*)(abase + k);
        }
    }

    if (w == 0) __builtin_amdgcn_s_wait_tensorcnt(0);
    __syncthreads();   // K tile 0 resident

    float runM[8], runS[8];
    #pragma unroll
    for (int e = 0; e < 8; ++e) { runM[e] = -1e30f; runS[e] = 0.0f; }

    for (int ms = 0; ms < 64; ++ms) {
        const int cur = ms & 1;
        // prefetch next K tile into the other buffer while we compute
        if (w == 0 && ms + 1 < 64)
            tdm_load_tile((unsigned)(uintptr_t)(void*)&lds_k[cur ^ 1][0],
                          (unsigned long long)(uintptr_t)
                              (kbase + (size_t)(ms + 1) * 64 * CDIM));

        const unsigned short* kt = &lds_k[cur][0];
        v8f s0 = v8f{0.f,0.f,0.f,0.f,0.f,0.f,0.f,0.f};
        v8f s1 = v8f{0.f,0.f,0.f,0.f,0.f,0.f,0.f,0.f};
        #pragma unroll
        for (int kc = 0; kc < 8; ++kc) {
            union { v16bf v; unsigned int u[8]; } b0, b1;
            // B = K^T: fragment row = K-tile row m, pairs along c
            const unsigned short* kb0 =
                &kt[(jw * 32 + ln) * QS + kc * 32 + (hi ? 16 : 0)];
            const unsigned short* kb1 = kb0 + 16 * QS;
            #pragma unroll
            for (int p = 0; p < 8; ++p) {
                b0.u[p] = *(const unsigned int*)(kb0 + 2 * p);
                b1.u[p] = *(const unsigned int*)(kb1 + 2 * p);
            }
            s0 = __builtin_amdgcn_wmma_f32_16x16x32_bf16(false, a[kc].v, false, b0.v, (short)0, s0, false, false);
            s1 = __builtin_amdgcn_wmma_f32_16x16x32_bf16(false, a[kc].v, false, b1.v, (short)0, s1, false, false);
        }

        // diagonal crossing: rows n0+16i+r hit col offset 16i+r, owned by
        // wave jw == i>>1, tile (i&1), lane ln == r, element e = r&7.
        if (ms == (int)blockIdx.x && jw == (i >> 1)) {
            #pragma unroll
            for (int e = 0; e < 8; ++e) {
                int r = e + (hi ? 8 : 0);
                float v = (i & 1) ? s1[e] : s0[e];
                if (ln == r) draw[i * 16 + r] = v;
            }
        }

        // online softmax update; rows live in 16-lane halves -> xor<16 shuffles
        #pragma unroll
        for (int e = 0; e < 8; ++e) {
            float v0 = s0[e], v1 = s1[e];
            float tm = fmaxf(v0, v1);
            #pragma unroll
            for (int m = 8; m >= 1; m >>= 1)
                tm = fmaxf(tm, __shfl_xor(tm, m, 32));
            float nM = fmaxf(runM[e], tm);
            float p = __expf(v0 - nM) + __expf(v1 - nM);
            #pragma unroll
            for (int m = 8; m >= 1; m >>= 1)
                p += __shfl_xor(p, m, 32);
            runS[e] = runS[e] * __expf(runM[e] - nM) + p;
            runM[e] = nM;
        }

        // tile ms+1 must have landed; all waves done reading buffer cur
        if (w == 0) __builtin_amdgcn_s_wait_tensorcnt(0);
        __syncthreads();
    }

    // publish per-(row, col-half) stats, combine, emit diagonal of softmax
    if (lane == 0 || lane == 16) {
        int roff = i * 16 + (hi ? 8 : 0);
        #pragma unroll
        for (int e = 0; e < 8; ++e) {
            wmaxs[jw][roff + e] = runM[e];
            wsums[jw][roff + e] = runS[e];
        }
    }
    __syncthreads();
    if (tid < 64) {
        float M0 = wmaxs[0][tid], M1 = wmaxs[1][tid];
        float M  = fmaxf(M0, M1);
        float S  = wsums[0][tid] * __expf(M0 - M) + wsums[1][tid] * __expf(M1 - M);
        // reference renormalizes by (1e-8 + row_sum) with row_sum == 1
        float dv = __expf(draw[tid] - M) / (S * (1.0f + 1e-8f));
        diag[(size_t)b * NDIM + n0 + tid] = dv;
    }
}

// ---------------------------------------------------------------------------
// Kernel 3: out[b,c,n] = X[b,c,n] * diag[b,n]   (streaming, float4)
// ---------------------------------------------------------------------------
__global__ __launch_bounds__(256) void scale_kernel(
    const float* __restrict__ X, const float* __restrict__ diag,
    float* __restrict__ out)
{
    constexpr size_t N4 = NDIM / 4;
    size_t i4 = (size_t)blockIdx.x * 256 + threadIdx.x;
    const float4 x = ((const float4*)X)[i4];
    size_t n4 = i4 % N4;
    size_t bc = i4 / N4;
    size_t bb = bc / CDIM;
    const float4 dg = ((const float4*)diag)[bb * N4 + n4];
    float4 o;
    o.x = x.x * dg.x; o.y = x.y * dg.y; o.z = x.z * dg.z; o.w = x.w * dg.w;
    ((float4*)out)[i4] = o;
}

extern "C" void kernel_launch(void* const* d_in, const int* in_sizes, int n_in,
                              void* d_out, int out_size, void* d_ws, size_t ws_size,
                              hipStream_t stream)
{
    const float* X  = (const float*)d_in[0];   // [B, C, N]
    const float* Wq = (const float*)d_in[1];   // [C, C]
    const float* Wk = (const float*)d_in[2];   // [C, C]
    float* out = (float*)d_out;                // [B, C, N]

    // workspace: Qb (16MB bf16) | Kb (16MB bf16) | diag (128KB f32)
    unsigned short* Qb = (unsigned short*)d_ws;
    unsigned short* Kb = Qb + (size_t)BDIM * NDIM * CDIM;
    float* diag = (float*)(Kb + (size_t)BDIM * NDIM * CDIM);

    dim3 grid(NDIM / 64, BDIM);
    proj_kernel<<<grid, 256, 0, stream>>>(X, Wq, Wk, Qb, Kb);
    attn_diag_kernel<<<grid, 256, 0, stream>>>(Qb, Kb, diag);
    size_t total4 = (size_t)BDIM * CDIM * NDIM / 4;
    scale_kernel<<<(unsigned)(total4 / 256), 256, 0, stream>>>(X, diag, out);
}
